// CausalSelfAttention_46712064311918
// MI455X (gfx1250) — compile-verified, matrix operand order checked by reading, not run on hardware
//
#include <hip/hip_runtime.h>
#include <hip/hip_bf16.h>

// ---------------------------------------------------------------------------
// CDNA5 (gfx1250) causal self-attention, f16 WMMA w/ f32 accumulate.
// B=1, T=2048, C=1024, H=16, D=64.
// ~34 GFLOP over ~33MB touched -> matrix-unit bound; all three GEMM stages run
// on v_wmma_f32_16x16x32_f16. 32x64 register tiles (8 WMMA per k-step, 12
// b128 loads) keep the WMMA pipes fed. Q strips staged via the Tensor Data
// Mover (tensor_load_to_lds + s_wait_tensorcnt) where the builtin exists.
// ---------------------------------------------------------------------------

typedef __attribute__((ext_vector_type(16))) _Float16 v16h;
typedef __attribute__((ext_vector_type(8)))  _Float16 v8h;
typedef __attribute__((ext_vector_type(8)))  float    v8f;

#if defined(__has_builtin)
#if __has_builtin(__builtin_amdgcn_tensor_load_to_lds) && __has_builtin(__builtin_amdgcn_s_wait_tensorcnt)
#define HAVE_TDM 1
#endif
#endif

__device__ __forceinline__ v16h cat8(v8h a, v8h b) {
  return __builtin_shufflevector(a, b, 0,1,2,3,4,5,6,7,8,9,10,11,12,13,14,15);
}

__device__ __forceinline__ v8f wmma_f16(v16h a, v16h b, v8f c) {
  // v_wmma_f32_16x16x32_f16 (neg_a, A, neg_b, B, c_mod, C, reuse_a, reuse_b)
  return __builtin_amdgcn_wmma_f32_16x16x32_f16(false, a, false, b, (short)0, c, false, false);
}

// A-style 16x32 f16 fragment from a row-major f16 matrix (row stride = ld).
// lane: row = row0 + (lane&15); K in [kb + h*8, +8) and [kb+16+h*8, +8).
__device__ __forceinline__ v16h load_a16(const _Float16* __restrict__ base,
                                         int ld, int row0, int kb, int lane) {
  int m = row0 + (lane & 15);
  int h = lane >> 4;
  const _Float16* p = base + (size_t)m * ld + kb + h * 8;
  v8h lo = *(const v8h*)(p);
  v8h hi = *(const v8h*)(p + 16);
  return cat8(lo, hi);
}

#ifdef HAVE_TDM
// TDM: DMA a 2D f16 tile (tile_h rows x tile_w elems, row stride = stride
// elems) from global memory into LDS at byte offset lds_off. Descriptor
// fields per CDNA5 ISA ch.8 (D# group0/group1).
__device__ __forceinline__ void tdm_load_2d_f16(unsigned lds_off,
                                                const void* gptr,
                                                int tile_w, int tile_h,
                                                int stride) {
  typedef __attribute__((ext_vector_type(4))) unsigned u32x4;
  typedef __attribute__((ext_vector_type(4))) int      i32x4;
  typedef __attribute__((ext_vector_type(8))) int      i32x8;
  unsigned long long ga = (unsigned long long)(uintptr_t)gptr;
  u32x4 g0;
  g0[0] = 1u;                                           // count=1 (user D#)
  g0[1] = lds_off;                                      // lds_addr (bytes)
  g0[2] = (unsigned)(ga & 0xFFFFFFFFu);                 // global_addr[31:0]
  g0[3] = (unsigned)((ga >> 32) & 0x1FFFFFFu) | (2u << 30); // addr[56:32]|type=2
  i32x8 g1;
  g1[0] = 1 << 16;                                      // data_size=2B
  g1[1] = (tile_w & 0xFFFF) << 16;                      // tensor_dim0 lo
  g1[2] = ((tile_w >> 16) & 0xFFFF) | ((tile_h & 0xFFFF) << 16); // dim0 hi|dim1 lo
  g1[3] = ((tile_h >> 16) & 0xFFFF) | ((tile_w & 0xFFFF) << 16); // dim1 hi|tile_dim0
  g1[4] = tile_h & 0xFFFF;                              // tile_dim1, tile_dim2=0
  g1[5] = stride;                                       // tensor_dim0_stride lo
  g1[6] = 0;
  g1[7] = 0;
  i32x4 z4 = {0, 0, 0, 0};
#if __clang_major__ >= 23
  i32x8 z8 = {0, 0, 0, 0, 0, 0, 0, 0};
  __builtin_amdgcn_tensor_load_to_lds(g0, g1, z4, z4, z8, 0);
#else
  __builtin_amdgcn_tensor_load_to_lds(g0, g1, z4, z4, 0);
#endif
}
#endif

// ---------------------------------------------------------------------------
// fp32 -> fp16 elementwise convert
// ---------------------------------------------------------------------------
__global__ void f32_to_f16_kernel(const float* __restrict__ src,
                                  _Float16* __restrict__ dst, int n) {
  int i = blockIdx.x * blockDim.x + threadIdx.x;
  if (i < n) dst[i] = (_Float16)src[i];
}

// ---------------------------------------------------------------------------
// Pack a K x N row-major f32 matrix into WMMA-B fragment order (f16) so the
// GEMM consumer does one contiguous 32B load per lane per WMMA.
// ---------------------------------------------------------------------------
__global__ void pack_b_kernel(const float* __restrict__ src,
                              _Float16* __restrict__ dst, int K, int N) {
  int i = blockIdx.x * blockDim.x + threadIdx.x;
  if (i >= K * N) return;
  int e    = i & 15;
  int lane = (i >> 4) & 31;
  int tile = i >> 9;
  int KT = K >> 5;
  int kt = tile % KT;
  int nt = tile / KT;
  int kk = kt * 32 + ((e >> 3) << 4) + ((lane >> 4) << 3) + (e & 7);
  int nn = nt * 16 + (lane & 15);
  dst[i] = (_Float16)src[(size_t)kk * N + nn];
}

// ---------------------------------------------------------------------------
// QKV GEMM (2048x1024 @ 1024x3072) + RoPE + scatter.
// One 32x64 output tile per wave: 2 A-frags x 4 B-frags -> 8 WMMAs per
// k-step, 12 b128 loads. Each wave's 64 columns = one head of one of q/k/v.
// ---------------------------------------------------------------------------
__global__ void qkv_rope_kernel(const _Float16* __restrict__ xh,
                                const _Float16* __restrict__ wpk,
                                _Float16* __restrict__ q,
                                _Float16* __restrict__ kd,
                                _Float16* __restrict__ vT) {
  const int KT = 32;                       // 1024 / 32
  int gw   = (blockIdx.x * blockDim.x + threadIdx.x) >> 5;
  int lane = threadIdx.x & 31;
  int mt = gw & 63;                        // 64 row tiles of 32 rows
  int ng = gw >> 6;                        // 48 col groups of 64 cols
  int h = lane >> 4, nl = lane & 15;

  v8f c[2][4] = {};
  const _Float16* arow0 = xh + (size_t)(mt * 32 + nl) * 1024 + h * 8;
  const _Float16* arow1 = arow0 + (size_t)16 * 1024;
  const _Float16* bptr  = wpk + ((size_t)(ng * 4) * KT * 32 + lane) * 16;

  #pragma unroll 2
  for (int kt = 0; kt < KT; ++kt) {
    v16h a0 = cat8(*(const v8h*)(arow0 + kt * 32),
                   *(const v8h*)(arow0 + kt * 32 + 16));
    v16h a1 = cat8(*(const v8h*)(arow1 + kt * 32),
                   *(const v8h*)(arow1 + kt * 32 + 16));
    #pragma unroll
    for (int j = 0; j < 4; ++j) {
      v16h b = *(const v16h*)(bptr + (size_t)j * 16384 + (size_t)kt * 512);
      c[0][j] = wmma_f16(a0, b, c[0][j]);
      c[1][j] = wmma_f16(a1, b, c[1][j]);
    }
  }

  int sect = (ng * 64) >> 10;              // 0=q 1=k 2=v (uniform per wave)
  int head = ng & 15;                      // uniform per wave

  if (sect == 2) {                         // V: store transposed [H][D][T]
    #pragma unroll
    for (int j = 0; j < 4; ++j) {
      _Float16* dst = vT + ((size_t)head * 64 + j * 16 + nl) * 2048;
      #pragma unroll
      for (int i = 0; i < 2; ++i)
        #pragma unroll
        for (int r = 0; r < 8; ++r)
          dst[mt * 32 + i * 16 + r + 8 * h] = (_Float16)c[i][j][r];
    }
  } else {                                 // Q/K: RoPE then [H][T][D]
    _Float16* base = (sect == 0 ? q : kd) + (size_t)head * 131072;
    #pragma unroll
    for (int j = 0; j < 4; ++j) {
      int d = j * 16 + nl;
      // inv_freq = 10000^(-(d>>1)/32) = exp(-(d>>1) * ln(10000)/32)
      float fr = __expf(-(float)(d >> 1) * 0.28782313662425572f);
      #pragma unroll
      for (int i = 0; i < 2; ++i) {
        #pragma unroll
        for (int r = 0; r < 8; ++r) {
          float xv = c[i][j][r];
          float pv = __shfl_xor(xv, 1, 32);  // partner column d^1 = lane^1
          int t = mt * 32 + i * 16 + r + 8 * h;
          float s, co;
          __sincosf((float)t * fr, &s, &co);
          float o = (d & 1) ? fmaf(pv, s, xv * co) : fmaf(xv, co, -pv * s);
          base[(size_t)t * 64 + d] = (_Float16)o;
        }
      }
    }
  }
}

// ---------------------------------------------------------------------------
// Flash attention: one wave = one (head, 16-row q strip). Q strip staged to
// LDS via TDM when available. Streams 32-key blocks: 4 WMMAs for scores
// (K=D=64), fp32 online softmax, probs bounce through per-wave LDS tile to
// A-fragment layout, 4 WMMAs for PV.
// ---------------------------------------------------------------------------
__global__ void flash_attn_kernel(const _Float16* __restrict__ q,
                                  const _Float16* __restrict__ k,
                                  const _Float16* __restrict__ vT,
                                  _Float16* __restrict__ O) {
  __shared__ __align__(16) _Float16 lds_p[4][512];    // 16x32 probs per wave
#ifdef HAVE_TDM
  __shared__ __align__(16) _Float16 lds_q[4][1024];   // 16x64 q per wave
#endif
  int wv = threadIdx.x >> 5;
  int lane = threadIdx.x & 31;
  int gw = blockIdx.x * 4 + wv;
  int head = gw >> 7;
  int qt   = gw & 127;
  int q0 = qt << 4;
  int h = lane >> 4, nl = lane & 15;

  const _Float16* qh = q  + (size_t)head * 131072;   // [T][D]
  const _Float16* kh = k  + (size_t)head * 131072;   // [T][D]
  const _Float16* vh = vT + (size_t)head * 131072;   // [D][T]

  v16h qf0, qf1;
#ifdef HAVE_TDM
  {
    unsigned loff = (unsigned)(uintptr_t)(&lds_q[wv][0]);  // LDS byte offset
    tdm_load_2d_f16(loff, qh + (size_t)q0 * 64, /*w=*/64, /*h=*/16, /*stride=*/64);
    __builtin_amdgcn_s_wait_tensorcnt(0);
    const _Float16* ql = &lds_q[wv][0];
    qf0 = cat8(*(const v8h*)(ql + nl * 64 + h * 8),
               *(const v8h*)(ql + nl * 64 + h * 8 + 16));
    qf1 = cat8(*(const v8h*)(ql + nl * 64 + 32 + h * 8),
               *(const v8h*)(ql + nl * 64 + 32 + h * 8 + 16));
  }
#else
  qf0 = load_a16(qh, 64, q0, 0, lane);
  qf1 = load_a16(qh, 64, q0, 32, lane);
#endif

  v8f o0 = {}, o1 = {}, o2 = {}, o3 = {};
  float mrow[8], lrow[8];
  #pragma unroll
  for (int r = 0; r < 8; ++r) { mrow[r] = -3.0e38f; lrow[r] = 0.f; }

  _Float16* plds = &lds_p[wv][0];
  const int kend = q0 + 16;

  for (int kb = 0; kb < kend; kb += 32) {
    if (kb + 32 < kend)
      __builtin_prefetch(kh + (size_t)(kb + 32 + nl) * 64 + h * 8, 0, 0);

    // scores: q(16x64) @ k^T  -> two 16x16 f32 tiles
    v8f s0 = {}, s1 = {};
    s0 = wmma_f16(qf0, load_a16(kh, 64, kb,      0,  lane), s0);
    s0 = wmma_f16(qf1, load_a16(kh, 64, kb,      32, lane), s0);
    s1 = wmma_f16(qf0, load_a16(kh, 64, kb + 16, 0,  lane), s1);
    s1 = wmma_f16(qf1, load_a16(kh, 64, kb + 16, 32, lane), s1);

    float a0[8], a1[8], alpha[8];
    #pragma unroll
    for (int r = 0; r < 8; ++r) {
      int qi = q0 + r + 8 * h;
      a0[r] = (kb + nl      <= qi) ? s0[r] * 0.125f : -3.0e38f;  // 1/sqrt(64)
      a1[r] = (kb + 16 + nl <= qi) ? s1[r] * 0.125f : -3.0e38f;
    }
    #pragma unroll
    for (int r = 0; r < 8; ++r) {
      float t = fmaxf(a0[r], a1[r]);
      t = fmaxf(t, __shfl_xor(t, 1, 32));
      t = fmaxf(t, __shfl_xor(t, 2, 32));
      t = fmaxf(t, __shfl_xor(t, 4, 32));
      t = fmaxf(t, __shfl_xor(t, 8, 32));
      float mnew = fmaxf(mrow[r], t);
      float al = __expf(mrow[r] - mnew);
      float p0 = __expf(a0[r] - mnew);
      float p1 = __expf(a1[r] - mnew);
      float ss = p0 + p1;
      ss += __shfl_xor(ss, 1, 32);
      ss += __shfl_xor(ss, 2, 32);
      ss += __shfl_xor(ss, 4, 32);
      ss += __shfl_xor(ss, 8, 32);
      lrow[r] = lrow[r] * al + ss;
      mrow[r] = mnew;
      alpha[r] = al;
      a0[r] = p0; a1[r] = p1;
    }
    #pragma unroll
    for (int r = 0; r < 8; ++r) {
      o0[r] *= alpha[r]; o1[r] *= alpha[r];
      o2[r] *= alpha[r]; o3[r] *= alpha[r];
    }
    // C-layout probs -> LDS [16 rows][32 keys] f16, re-read in A layout
    #pragma unroll
    for (int r = 0; r < 8; ++r) {
      int m = r + 8 * h;
      plds[m * 32 + nl]      = (_Float16)a0[r];
      plds[m * 32 + 16 + nl] = (_Float16)a1[r];
    }
    asm volatile("s_wait_dscnt 0" ::: "memory");   // LDS RAW within wave
    v16h pf = cat8(*(const v8h*)(plds + nl * 32 + h * 8),
                   *(const v8h*)(plds + nl * 32 + 16 + h * 8));

    // O += P(16x32) @ V(32x64); vT rows are d -> row-contiguous B fragments
    o0 = wmma_f16(pf, load_a16(vh, 2048, 0,  kb, lane), o0);
    o1 = wmma_f16(pf, load_a16(vh, 2048, 16, kb, lane), o1);
    o2 = wmma_f16(pf, load_a16(vh, 2048, 32, kb, lane), o2);
    o3 = wmma_f16(pf, load_a16(vh, 2048, 48, kb, lane), o3);
  }

  #pragma unroll
  for (int r = 0; r < 8; ++r) {
    int t = q0 + r + 8 * h;
    float inv = 1.0f / lrow[r];
    size_t base = (size_t)t * 1024 + head * 64 + nl;
    O[base]      = (_Float16)(o0[r] * inv);
    O[base + 16] = (_Float16)(o1[r] * inv);
    O[base + 32] = (_Float16)(o2[r] * inv);
    O[base + 48] = (_Float16)(o3[r] * inv);
  }
}

// ---------------------------------------------------------------------------
// Output projection: O(2048x1024 f16) @ w_proj -> d_out f32. 32x64 tiles.
// ---------------------------------------------------------------------------
__global__ void proj_kernel(const _Float16* __restrict__ Oh,
                            const _Float16* __restrict__ wpk,
                            float* __restrict__ out) {
  const int KT = 32;
  int gw   = (blockIdx.x * blockDim.x + threadIdx.x) >> 5;
  int lane = threadIdx.x & 31;
  int mt = gw & 63;                        // 64 row tiles of 32
  int ng = gw >> 6;                        // 16 col groups of 64
  int h = lane >> 4, nl = lane & 15;

  v8f c[2][4] = {};
  const _Float16* arow0 = Oh + (size_t)(mt * 32 + nl) * 1024 + h * 8;
  const _Float16* arow1 = arow0 + (size_t)16 * 1024;
  const _Float16* bptr  = wpk + ((size_t)(ng * 4) * KT * 32 + lane) * 16;

  #pragma unroll 2
  for (int kt = 0; kt < KT; ++kt) {
    v16h a0 = cat8(*(const v8h*)(arow0 + kt * 32),
                   *(const v8h*)(arow0 + kt * 32 + 16));
    v16h a1 = cat8(*(const v8h*)(arow1 + kt * 32),
                   *(const v8h*)(arow1 + kt * 32 + 16));
    #pragma unroll
    for (int j = 0; j < 4; ++j) {
      v16h b = *(const v16h*)(bptr + (size_t)j * 16384 + (size_t)kt * 512);
      c[0][j] = wmma_f16(a0, b, c[0][j]);
      c[1][j] = wmma_f16(a1, b, c[1][j]);
    }
  }
  #pragma unroll
  for (int i = 0; i < 2; ++i)
    #pragma unroll
    for (int j = 0; j < 4; ++j)
      #pragma unroll
      for (int r = 0; r < 8; ++r) {
        int t = mt * 32 + i * 16 + r + 8 * h;
        out[(size_t)t * 1024 + ng * 64 + j * 16 + nl] = c[i][j][r];
      }
}

// ---------------------------------------------------------------------------
extern "C" void kernel_launch(void* const* d_in, const int* in_sizes, int n_in,
                              void* d_out, int out_size, void* d_ws, size_t ws_size,
                              hipStream_t stream) {
  const float* x     = (const float*)d_in[0];   // (1,2048,1024)
  const float* wqkv  = (const float*)d_in[1];   // (1024,3072)
  const float* wproj = (const float*)d_in[2];   // (1024,1024)
  float* out = (float*)d_out;                   // (1,2048,1024)

  char* ws = (char*)d_ws;                       // needs ~28 MB
  _Float16* xh       = (_Float16*)(ws);                         // 4 MB
  _Float16* wqkv_pk  = (_Float16*)(ws + ((size_t)4  << 20));    // 6 MB
  _Float16* wproj_pk = (_Float16*)(ws + ((size_t)10 << 20));    // 2 MB
  _Float16* q        = (_Float16*)(ws + ((size_t)12 << 20));    // 4 MB
  _Float16* k        = (_Float16*)(ws + ((size_t)16 << 20));    // 4 MB
  _Float16* vT       = (_Float16*)(ws + ((size_t)20 << 20));    // 4 MB
  _Float16* O        = (_Float16*)(ws + ((size_t)24 << 20));    // 4 MB

  f32_to_f16_kernel<<<2097152 / 256, 256, 0, stream>>>(x, xh, 2097152);
  pack_b_kernel<<<3145728 / 256, 256, 0, stream>>>(wqkv, wqkv_pk, 1024, 3072);
  pack_b_kernel<<<1048576 / 256, 256, 0, stream>>>(wproj, wproj_pk, 1024, 1024);

  // 64x48 tiles of 32x64, 1 tile/wave, 8 waves/block
  qkv_rope_kernel<<<384, 256, 0, stream>>>(xh, wqkv_pk, q, k, vT);
  // 16 heads x 128 q-strips = 2048 waves, 4 waves/block
  flash_attn_kernel<<<512, 128, 0, stream>>>(q, k, vT, O);
  // 64x16 tiles of 32x64, 8 waves/block
  proj_kernel<<<128, 256, 0, stream>>>(O, wproj_pk, out);
}
